// MyLSTMCell_3169685865216
// MI455X (gfx1250) — matrix operand order repaired and checked by reading, not compiled, not run on hardware
//
#include <hip/hip_runtime.h>

typedef __bf16 bf16;
typedef __bf16 bf16x4  __attribute__((ext_vector_type(4)));
typedef __bf16 bf16x8  __attribute__((ext_vector_type(8)));
typedef __bf16 bf16x16 __attribute__((ext_vector_type(16)));
typedef float  f32x4   __attribute__((ext_vector_type(4)));
typedef float  f32x8   __attribute__((ext_vector_type(8)));
typedef unsigned int u32x4 __attribute__((ext_vector_type(4)));
typedef int          i32x4 __attribute__((ext_vector_type(4)));
typedef int          i32x8 __attribute__((ext_vector_type(8)));

#define B_SZ   8192
#define IN_SZ  1024
#define H_SZ   1024
#define K_SZ   2048          // fused K: [x | h]
#define NG     4096          // 4 gates * H rows of fused weight matrix
#define TM     64            // M tile (batch rows) per workgroup
#define TO     32            // o columns per workgroup (covers all 4 gates)
#define KT     32            // k step == WMMA bf16 K
#define NSTEPS (K_SZ / KT)   // 64
#define APAD   40            // padded LDS row stride (80B): 64B row + 16B pad

#if __has_builtin(__builtin_amdgcn_tensor_load_to_lds)
#define HAS_TDM 1
#else
#define HAS_TDM 0
#endif

__device__ __forceinline__ float sigf(float x) {
    return 1.0f / (1.0f + __expf(-x));
}
__device__ __forceinline__ float tanh_fast(float x) {
    return 1.0f - 2.0f / (__expf(2.0f * x) + 1.0f);   // saturates correctly
}

// ---- prep: fp32 -> bf16 conversions into workspace ----

__global__ void cvt_xh_kernel(const float* __restrict__ x,
                              const float* __restrict__ h,
                              bf16* __restrict__ xh) {
    size_t e = ((size_t)blockIdx.x * 256 + threadIdx.x) * 4;
    int b = (int)(e >> 11);
    int k = (int)(e & 2047);
    const float* src = (k < IN_SZ) ? (x + (size_t)b * IN_SZ + k)
                                   : (h + (size_t)b * H_SZ + (k - IN_SZ));
    f32x4 v = *(const f32x4*)src;
    bf16x4 o;
    o[0] = (bf16)v[0]; o[1] = (bf16)v[1]; o[2] = (bf16)v[2]; o[3] = (bf16)v[3];
    *(bf16x4*)(xh + e) = o;
}

__global__ void cvt_w_kernel(const float* __restrict__ Wx,
                             const float* __restrict__ Wh,
                             bf16* __restrict__ w) {
    size_t e = ((size_t)blockIdx.x * 256 + threadIdx.x) * 4;
    int r = (int)(e >> 11);          // fused row = g*1024 + o (Wx flat = [4096][1024])
    int k = (int)(e & 2047);
    const float* src = (k < IN_SZ) ? (Wx + (size_t)r * IN_SZ + k)
                                   : (Wh + (size_t)r * H_SZ + (k - IN_SZ));
    f32x4 v = *(const f32x4*)src;
    bf16x4 o;
    o[0] = (bf16)v[0]; o[1] = (bf16)v[1]; o[2] = (bf16)v[2]; o[3] = (bf16)v[3];
    *(bf16x4*)(w + e) = o;
}

__global__ void cvt_bias_kernel(const float* __restrict__ bx,
                                const float* __restrict__ bh,
                                float* __restrict__ bsum) {
    int i = blockIdx.x * 256 + threadIdx.x;
    bsum[i] = bx[i] + bh[i];
}

#if HAS_TDM
// Build a TDM D# (data_size = 4B units, LDS padding: +4 DWORDs per 16 DWORDs
// => 80B padded rows matching APAD) and issue one tensor_load_to_lds.
// 2-D tensors pass tile2 = 0 / tdim2 = 0; 3-D sets tile2/tdim2 and stride1.
__device__ __forceinline__ void tdm_tile_load(
    unsigned int lds_addr, unsigned long long gaddr,
    unsigned int td0, unsigned int td1,
    unsigned int tile0, unsigned int tile1, unsigned int tile2,
    unsigned long long stride0, unsigned long long stride1,
    unsigned int tdim2)
{
    u32x4 g0;
    g0[0] = 1u;                                       // count=1, user mode
    g0[1] = lds_addr;                                 // LDS byte address
    g0[2] = (unsigned int)gaddr;                      // global_addr[31:0]
    g0[3] = ((unsigned int)(gaddr >> 32) & 0x01FFFFFFu) | (2u << 30); // addr[56:32] | type=2

    i32x8 g1;
    g1[0] = (int)((2u << 16)      // data_size = 4B
                | (1u << 20)      // pad_enable
                | (3u << 22)      // pad_interval: 16 DWORDs (64B row)
                | (3u << 25));    // pad_amount:   4 DWORDs (16B)
    g1[1] = (int)((td0 & 0xFFFFu) << 16);                       // tensor_dim0[15:0]
    g1[2] = (int)((td0 >> 16) | ((td1 & 0xFFFFu) << 16));       // td0[31:16] | td1[15:0]
    g1[3] = (int)((td1 >> 16) | (tile0 << 16));                 // td1[31:16] | tile_dim0
    g1[4] = (int)(tile1 | (tile2 << 16));                       // tile_dim1 | tile_dim2
    g1[5] = (int)(unsigned int)stride0;                         // tensor_dim0_stride[31:0]
    g1[6] = (int)(((unsigned int)(stride0 >> 32) & 0xFFFFu)
                | (((unsigned int)stride1 & 0xFFFFu) << 16));   // td0s[47:32] | td1s[15:0]
    g1[7] = (int)(unsigned int)(stride1 >> 16);                 // td1s[47:16]

    i32x4 g2;
    g2[0] = (int)tdim2;   // tensor_dim2 (0 for 2-D)
    g2[1] = 0;            // tensor_dim3
    g2[2] = 0;            // tensor_dim2_stride[31:0] (4th dim, unused)
    g2[3] = 0;            // td2s[47:32] | tile_dim3
    i32x4 g3 = {0, 0, 0, 0};
    i32x8 gx = {0, 0, 0, 0, 0, 0, 0, 0};   // 2nd-descriptor slot; unused (count=1)

    __builtin_amdgcn_tensor_load_to_lds(g0, g1, g2, g3, gx, 0);
}
#endif

// ---- fused GEMM (bf16 WMMA) + LSTM epilogue ----
// Workgroup: 64 batch rows x 32 o-cols x 4 gates. 8 waves = (2 M-halves) x (4 gates).
// Each wave: 2x2 grid of 16x16 WMMA tiles for its gate, K accumulated over 2048.

__global__ __launch_bounds__(256) void lstm_gemm_kernel(
    const bf16*  __restrict__ xh,    // [8192][2048]
    const bf16*  __restrict__ w,     // [4096][2048], row = g*1024 + o
    const float* __restrict__ bsum,  // [4096]
    const float* __restrict__ cin,   // [8192][1024]
    float* __restrict__ hout,        // [8192][1024]
    float* __restrict__ cout)        // [8192][1024]
{
    __shared__ bf16  Alds[2][TM][APAD];       // 10240 B (TDM-padded rows)
    __shared__ bf16  Blds[2][128][APAD];      // 20480 B (4 gates * 32 o rows)
    __shared__ float act[4][TM][TO];          // 32768 B gate pre-activations

    const int tid   = threadIdx.x;
    const int lane  = tid & 31;
    const int wave  = tid >> 5;
    const int wgate = wave & 3;
    const int wm    = wave >> 2;
    const int mTile = blockIdx.y * TM;
    const int oTile = blockIdx.x * TO;

    f32x8 acc[2][2] = {};

    // ---- WMMA fragment LDS addressing (per CDNA5 ISA layouts) ----
    const int halfsel = lane >> 4;
    const int l15     = lane & 15;
    const int a_ko    = halfsel * 8;          // A: K 0..7 / 8..15 (+16 for hi chunk)
    const int b_ko    = halfsel * 8;          // B: K 0..7 / 8..15 (+16 for hi chunk)

    auto do_compute = [&](int cur) {
        bf16x16 afrag[2];
        bf16x16 bfrag[2];
        #pragma unroll
        for (int mi = 0; mi < 2; ++mi) {
            const int row = wm * 32 + mi * 16 + l15;
            bf16x8 lo = *(const bf16x8*)&Alds[cur][row][a_ko];
            bf16x8 hi = *(const bf16x8*)&Alds[cur][row][a_ko + 16];
            #pragma unroll
            for (int i2 = 0; i2 < 8; ++i2) { afrag[mi][i2] = lo[i2]; afrag[mi][i2 + 8] = hi[i2]; }
        }
        #pragma unroll
        for (int ni = 0; ni < 2; ++ni) {
            const int brow = wgate * 32 + ni * 16 + l15;
            bf16x8 lo = *(const bf16x8*)&Blds[cur][brow][b_ko];
            bf16x8 hi = *(const bf16x8*)&Blds[cur][brow][b_ko + 16];
            #pragma unroll
            for (int i2 = 0; i2 < 8; ++i2) { bfrag[ni][i2] = lo[i2]; bfrag[ni][i2 + 8] = hi[i2]; }
        }
        #pragma unroll
        for (int mi = 0; mi < 2; ++mi)
            #pragma unroll
            for (int ni = 0; ni < 2; ++ni)
                acc[mi][ni] = __builtin_amdgcn_wmma_f32_16x16x32_bf16(
                    false, afrag[mi], false, bfrag[ni],
                    (short)0, acc[mi][ni], false, false);
    };

#if HAS_TDM
    // ---- Tensor Data Mover pipeline: wave 0 issues 2 descriptors per k-step ----
    const unsigned int a_lds[2] = {
        (unsigned int)(size_t)&Alds[0][0][0], (unsigned int)(size_t)&Alds[1][0][0] };
    const unsigned int b_lds[2] = {
        (unsigned int)(size_t)&Blds[0][0][0], (unsigned int)(size_t)&Blds[1][0][0] };
    const unsigned long long a_gbase =
        (unsigned long long)(size_t)xh + (unsigned long long)mTile * K_SZ * 2;
    const unsigned long long b_gbase =
        (unsigned long long)(size_t)w + (unsigned long long)oTile * K_SZ * 2;

    if (wave == 0) {
        // prologue: buffer 0 <- k-step 0
        tdm_tile_load(a_lds[0], a_gbase, 16, TM, 16, TM, 0, 1024ull, 0ull, 0);
        tdm_tile_load(b_lds[0], b_gbase, 16, 32, 16, 32, 4,
                      1024ull, (unsigned long long)H_SZ * (K_SZ / 2), 4);
    }

    for (int kt = 0; kt < NSTEPS; ++kt) {
        __syncthreads();                       // all waves done reading buf[(kt+1)&1]
        const int cur = kt & 1;
        if (wave == 0) {
            if (kt + 1 < NSTEPS) {
                const int nxt = cur ^ 1;
                const unsigned long long koffb = (unsigned long long)(kt + 1) * KT * 2;
                tdm_tile_load(a_lds[nxt], a_gbase + koffb, 16, TM, 16, TM, 0, 1024ull, 0ull, 0);
                tdm_tile_load(b_lds[nxt], b_gbase + koffb, 16, 32, 16, 32, 4,
                              1024ull, (unsigned long long)H_SZ * (K_SZ / 2), 4);
                __builtin_amdgcn_s_wait_tensorcnt(2);   // buf cur complete (in-order)
            } else {
                __builtin_amdgcn_s_wait_tensorcnt(0);
            }
        }
        __syncthreads();                       // buf cur visible to all waves
        do_compute(cur);
    }
#else
    // ---- fallback: register-staged global->LDS double buffering ----
    const int arow = tid >> 2;
    const int akc  = (tid & 3) * 8;
    const size_t a_gbase = (size_t)(mTile + arow) * K_SZ + akc;
    const int br0 = tid >> 2;
    const int br1 = 64 + (tid >> 2);
    const int g0r = br0 >> 5, j0 = br0 & 31;
    const int g1r = br1 >> 5, j1 = br1 & 31;
    const size_t b_g0 = (size_t)(g0r * H_SZ + oTile + j0) * K_SZ + akc;
    const size_t b_g1 = (size_t)(g1r * H_SZ + oTile + j1) * K_SZ + akc;
    {
        bf16x8 a  = *(const bf16x8*)(xh + a_gbase);
        bf16x8 b0 = *(const bf16x8*)(w + b_g0);
        bf16x8 b1 = *(const bf16x8*)(w + b_g1);
        *(bf16x8*)&Alds[0][arow][akc] = a;
        *(bf16x8*)&Blds[0][br0][akc]  = b0;
        *(bf16x8*)&Blds[0][br1][akc]  = b1;
    }
    for (int kt = 0; kt < NSTEPS; ++kt) {
        __syncthreads();
        const int cur  = kt & 1;
        const bool more = (kt + 1) < NSTEPS;
        bf16x8 an, bn0, bn1;
        if (more) {
            const size_t koff = (size_t)(kt + 1) * KT;
            an  = *(const bf16x8*)(xh + a_gbase + koff);
            bn0 = *(const bf16x8*)(w + b_g0 + koff);
            bn1 = *(const bf16x8*)(w + b_g1 + koff);
        }
        do_compute(cur);
        if (more) {
            const int nxt = cur ^ 1;
            *(bf16x8*)&Alds[nxt][arow][akc] = an;
            *(bf16x8*)&Blds[nxt][br0][akc]  = bn0;
            *(bf16x8*)&Blds[nxt][br1][akc]  = bn1;
        }
    }
#endif

    // ---- stage gate pre-activations to LDS (C layout: vgpr r -> M = base + r + 8*half) ----
    #pragma unroll
    for (int mi = 0; mi < 2; ++mi)
        #pragma unroll
        for (int ni = 0; ni < 2; ++ni)
            #pragma unroll
            for (int r = 0; r < 8; ++r) {
                const int row = wm * 32 + mi * 16 + halfsel * 8 + r;
                const int col = ni * 16 + l15;
                act[wgate][row][col] = acc[mi][ni][r];
            }
    __syncthreads();

    // ---- LSTM epilogue: 2048 (b,o) cells, 8 per thread ----
    #pragma unroll
    for (int j = 0; j < 8; ++j) {
        const int e   = tid + j * 256;
        const int row = e >> 5;
        const int col = e & 31;
        const int b   = mTile + row;
        const int o   = oTile + col;
        const float gi = act[0][row][col] + bsum[0 * H_SZ + o];
        const float gf = act[1][row][col] + bsum[1 * H_SZ + o];
        const float gg = act[2][row][col] + bsum[2 * H_SZ + o];
        const float go = act[3][row][col] + bsum[3 * H_SZ + o];
        const float iv = sigf(gi);
        const float fv = sigf(gf);
        const float gv = tanh_fast(gg);
        const float ov = sigf(go);
        const float cold = cin[(size_t)b * H_SZ + o];
        const float cnew = fv * cold + iv * gv;
        const float hnew = ov * tanh_fast(cnew);
        cout[(size_t)b * H_SZ + o] = cnew;
        hout[(size_t)b * H_SZ + o] = hnew;
    }
}

extern "C" void kernel_launch(void* const* d_in, const int* in_sizes, int n_in,
                              void* d_out, int out_size, void* d_ws, size_t ws_size,
                              hipStream_t stream) {
    const float* x  = (const float*)d_in[0];
    const float* h  = (const float*)d_in[1];
    const float* c  = (const float*)d_in[2];
    const float* Wx = (const float*)d_in[3];
    const float* bx = (const float*)d_in[4];
    const float* Wh = (const float*)d_in[5];
    const float* bh = (const float*)d_in[6];

    bf16*  xh   = (bf16*)d_ws;                               // 32 MiB
    bf16*  w    = xh + (size_t)B_SZ * K_SZ;                  // 16 MiB
    float* bsum = (float*)(w + (size_t)NG * K_SZ);           // 16 KiB

    float* hout = (float*)d_out;
    float* cout = hout + (size_t)B_SZ * H_SZ;

    cvt_xh_kernel<<<(B_SZ * K_SZ) / (256 * 4), 256, 0, stream>>>(x, h, xh);
    cvt_w_kernel<<<(NG * K_SZ) / (256 * 4), 256, 0, stream>>>(Wx, Wh, w);
    cvt_bias_kernel<<<NG / 256, 256, 0, stream>>>(bx, bh, bsum);

    dim3 grid(H_SZ / TO, B_SZ / TM);   // (32, 128)
    lstm_gemm_kernel<<<grid, 256, 0, stream>>>(xh, w, bsum, c, hout, cout);
}
